// MambaBlock_17239998726182
// MI455X (gfx1250) — matrix operand order repaired and checked by reading, not compiled, not run on hardware
//
#include <hip/hip_runtime.h>
#include <hip/hip_bf16.h>

// ---------------------------------------------------------------------------
// Mamba block for MI455X (gfx1250, wave32).
// GEMMs run on v_wmma_f32_16x16x32_bf16; scan runs on VALU with LDS broadcast.
// ---------------------------------------------------------------------------

#define Bq   2
#define Lq   1024
#define DMq  1024
#define DINq 2048
#define NST  16      // DSTATE
#define DCV  4       // DCONV
#define DTR  64      // DTRANK
#define MROWS (Bq * Lq)          // 2048 token rows

typedef unsigned short u16;
typedef __attribute__((ext_vector_type(8)))  unsigned short v8u;
typedef __attribute__((ext_vector_type(16))) unsigned short v16u;
typedef __attribute__((ext_vector_type(16))) __bf16         v16bf;
typedef __attribute__((ext_vector_type(8)))  float          v8f;

__device__ __forceinline__ u16 f2bf(float f) {
    unsigned int u = __float_as_uint(f);
    unsigned int r = u + 0x7FFFu + ((u >> 16) & 1u);   // round-to-nearest-even
    return (u16)(r >> 16);
}

__device__ __forceinline__ float silu(float v) {
    return v / (1.f + __expf(-v));
}

// ---------------------------------------------------------------------------
// f32 -> bf16 raw conversion (weights)
// ---------------------------------------------------------------------------
__global__ __launch_bounds__(256) void k_f32_to_bf16(const float* __restrict__ src,
                                                     u16* __restrict__ dst, int n) {
    int i = blockIdx.x * 256 + threadIdx.x;
    if (i < n) dst[i] = f2bf(src[i]);
}

// ---------------------------------------------------------------------------
// LayerNorm over DM=1024 per token row; emits bf16 activations.
// ---------------------------------------------------------------------------
__global__ __launch_bounds__(256) void k_layernorm_bf16(const float* __restrict__ x,
                                                        const float* __restrict__ w,
                                                        const float* __restrict__ b,
                                                        u16* __restrict__ out) {
    __shared__ float s1[256];
    __shared__ float s2[256];
    const int row = blockIdx.x;                       // 0..MROWS-1
    const float* xr = x + (size_t)row * DMq;
    float sum = 0.f, sq = 0.f;
    for (int i = threadIdx.x; i < DMq; i += 256) {
        float v = xr[i];
        sum += v; sq += v * v;
    }
    s1[threadIdx.x] = sum; s2[threadIdx.x] = sq;
    __syncthreads();
    for (int s = 128; s > 0; s >>= 1) {
        if (threadIdx.x < s) {
            s1[threadIdx.x] += s1[threadIdx.x + s];
            s2[threadIdx.x] += s2[threadIdx.x + s];
        }
        __syncthreads();
    }
    const float mu  = s1[0] * (1.f / DMq);
    const float var = s2[0] * (1.f / DMq) - mu * mu;
    const float rs  = rsqrtf(var + 1e-5f);
    for (int i = threadIdx.x; i < DMq; i += 256) {
        float v = (xr[i] - mu) * rs * w[i] + b[i];
        out[(size_t)row * DMq + i] = f2bf(v);
    }
}

// ---------------------------------------------------------------------------
// Generic bf16 GEMM:  C[M,N] (f32) = A[M,K] * W[N,K]^T   (+ optional addsrc)
// Block: 256 threads = 8 waves; block tile 64(M) x 128(N);
// each wave owns a 16x64 strip = 4x v_wmma_f32_16x16x32_bf16 accumulators.
// Fragment loads use one contiguous k order, the same for A(16x32) and B(32x16):
//   A: lane(m=lane&15, hi=lane>>4) holds K = 16*hi+{0..15}
//   B: lane(n=lane&15, hi=lane>>4) holds K = 16*hi+{0..15}
// Both are contiguous chunks in K -> plain b128 loads, L2-resident.
// ---------------------------------------------------------------------------
__global__ __launch_bounds__(256) void k_gemm_bf16_wmma(const u16* __restrict__ A,
                                                        const u16* __restrict__ W,
                                                        float* __restrict__ C,
                                                        int M, int N, int K,
                                                        const float* __restrict__ addsrc) {
    const int tid  = threadIdx.x;
    const int wave = tid >> 5;
    const int lane = tid & 31;
    const int l15  = lane & 15;
    const int hi   = lane >> 4;
    const int m0   = blockIdx.y * 64  + (wave & 3) * 16;
    const int n0   = blockIdx.x * 128 + (wave >> 2) * 64;

    v8f acc[4] = {};

    int mA = m0 + l15; if (mA >= M) mA = M - 1;
    const u16* arow = A + (size_t)mA * K;
    const u16* wrow[4];
#pragma unroll
    for (int t = 0; t < 4; ++t) {
        int n = n0 + t * 16 + l15; if (n >= N) n = N - 1;
        wrow[t] = W + (size_t)n * K;
    }

    for (int k0 = 0; k0 < K; k0 += 32) {
        union { v16u u; v16bf b; } af;
        union { v16u u; v16bf b; } bf[4];
        af.u = *(const v16u*)(arow + k0 + 16 * hi);
#pragma unroll
        for (int t = 0; t < 4; ++t)
            bf[t].u = *(const v16u*)(wrow[t] + k0 + 16 * hi);
#pragma unroll
        for (int t = 0; t < 4; ++t)
            acc[t] = __builtin_amdgcn_wmma_f32_16x16x32_bf16(
                false, af.b, false, bf[t].b, (short)0, acc[t], false, false);
    }

    // C/D layout: VGPR r -> row m0 + 8*hi + r, column n0 + t*16 + (lane&15)
#pragma unroll
    for (int t = 0; t < 4; ++t) {
        int n = n0 + t * 16 + l15;
        if (n >= N) continue;
#pragma unroll
        for (int r = 0; r < 8; ++r) {
            int m = m0 + hi * 8 + r;
            if (m < M) {
                size_t idx = (size_t)m * N + n;
                float v = acc[t][r];
                if (addsrc) v += addsrc[idx];
                C[idx] = v;
            }
        }
    }
}

// ---------------------------------------------------------------------------
// Causal depthwise conv1d (DCONV=4) + bias + SiLU on xm = xz[..., :DIN].
// Emits f32 (for scan) and bf16 (for x_proj GEMM).
// ---------------------------------------------------------------------------
__global__ __launch_bounds__(256) void k_conv_silu(const float* __restrict__ xz,
                                                   const float* __restrict__ cw,
                                                   const float* __restrict__ cb,
                                                   float* __restrict__ xc,
                                                   u16* __restrict__ xcbf) {
    const int idx = blockIdx.x * 256 + threadIdx.x;      // B*L*DIN = 2^22
    if (idx >= Bq * Lq * DINq) return;
    const int d = idx & (DINq - 1);
    const int l = (idx >> 11) & (Lq - 1);
    const int b = idx >> 21;
    float acc = cb[d];
    const float* base = xz + (size_t)b * Lq * (2 * DINq) + d;
#pragma unroll
    for (int j = 0; j < DCV; ++j) {
        int t = l - (DCV - 1) + j;
        if (t >= 0) acc += cw[d * DCV + j] * base[(size_t)t * (2 * DINq)];
    }
    float s = silu(acc);
    xc[idx]   = s;
    xcbf[idx] = f2bf(s);
}

// ---------------------------------------------------------------------------
// Extract dt-rank columns (first 64 of the 96-wide x_dbl rows) as bf16.
// ---------------------------------------------------------------------------
__global__ __launch_bounds__(256) void k_extract_dt(const float* __restrict__ xdbl,
                                                    u16* __restrict__ out) {
    int idx = blockIdx.x * 256 + threadIdx.x;            // MROWS*64
    if (idx >= MROWS * DTR) return;
    int r = idx & (DTR - 1);
    int m = idx >> 6;
    out[idx] = f2bf(xdbl[(size_t)m * (DTR + 2 * NST) + r]);
}

// ---------------------------------------------------------------------------
// dt = softplus(dt_pre + bias)  (in place)
// ---------------------------------------------------------------------------
__global__ __launch_bounds__(256) void k_dt_softplus(float* __restrict__ dt,
                                                     const float* __restrict__ bias) {
    int idx = blockIdx.x * 256 + threadIdx.x;            // MROWS*DIN
    if (idx >= MROWS * DINq) return;
    float v = dt[idx] + bias[idx & (DINq - 1)];
    dt[idx] = (v > 20.f) ? v : log1pf(__expf(v));
}

// ---------------------------------------------------------------------------
// Selective scan: one lane per (b,d); 16 recurrent states in VGPRs.
// Per-timestep B/C (16+16 floats) broadcast through LDS.
// Fused epilogue: y = (scan + xc*D) * silu(z) -> bf16 for out_proj.
// ---------------------------------------------------------------------------
__global__ __launch_bounds__(256) void k_selective_scan(const float* __restrict__ xdbl,
                                                        const float* __restrict__ dtb,
                                                        const float* __restrict__ xc,
                                                        const float* __restrict__ xz,
                                                        const float* __restrict__ A_log,
                                                        const float* __restrict__ D_skip,
                                                        u16* __restrict__ ybf) {
    __shared__ float sB[NST];
    __shared__ float sC[NST];
    const int d = blockIdx.x * 256 + threadIdx.x;        // channel
    const int b = blockIdx.y;
    float Aneg[NST], h[NST];
#pragma unroll
    for (int n = 0; n < NST; ++n) {
        Aneg[n] = -__expf(A_log[(size_t)d * NST + n]);
        h[n] = 0.f;
    }
    const float Dv = D_skip[d];
    for (int l = 0; l < Lq; ++l) {
        const size_t row = (size_t)(b * Lq + l);
        if (threadIdx.x < 2 * NST)                       // row layout: [dt(64) | B(16) | C(16)]
            (&sB[0])[threadIdx.x] = xdbl[row * (DTR + 2 * NST) + DTR + threadIdx.x];
        __syncthreads();
        const float dtv = dtb[row * DINq + d];
        const float xcv = xc[row * DINq + d];
        const float dbu = dtv * xcv;
        float acc = 0.f;
#pragma unroll
        for (int n = 0; n < NST; ++n) {
            float dA = __expf(dtv * Aneg[n]);
            h[n] = dA * h[n] + dbu * sB[n];
            acc += h[n] * sC[n];
        }
        const float zv = xz[row * (2 * DINq) + DINq + d];
        const float y = (acc + xcv * Dv) * silu(zv);
        ybf[row * DINq + d] = f2bf(y);
        __syncthreads();
    }
}

// ---------------------------------------------------------------------------
// Launch
// ---------------------------------------------------------------------------
extern "C" void kernel_launch(void* const* d_in, const int* in_sizes, int n_in,
                              void* d_out, int out_size, void* d_ws, size_t ws_size,
                              hipStream_t stream) {
    (void)in_sizes; (void)n_in; (void)out_size; (void)ws_size;

    const float* x        = (const float*)d_in[0];   // [B,L,DM]
    const float* norm_w   = (const float*)d_in[1];
    const float* norm_b   = (const float*)d_in[2];
    const float* in_w     = (const float*)d_in[3];   // [2*DIN, DM]
    const float* conv_w   = (const float*)d_in[4];   // [DIN,1,4]
    const float* conv_b   = (const float*)d_in[5];
    const float* xp_w     = (const float*)d_in[6];   // [96, DIN]
    const float* dt_w     = (const float*)d_in[7];   // [DIN, 64]
    const float* dt_b     = (const float*)d_in[8];
    const float* A_log    = (const float*)d_in[9];   // [DIN, 16]
    const float* D_skip   = (const float*)d_in[10];
    const float* out_w    = (const float*)d_in[11];  // [DM, DIN]
    float* out            = (float*)d_out;           // [B,L,DM]

    // workspace carve-up (all sizes 256B-aligned by construction)
    char* ws = (char*)d_ws;
    size_t off = 0;
    auto take = [&](size_t bytes) { char* p = ws + off; off += bytes; return p; };
    u16*   h_bf    = (u16*)  take((size_t)MROWS * DMq * 2);            //  4 MB
    u16*   w_in    = (u16*)  take((size_t)2 * DINq * DMq * 2);         //  8 MB
    u16*   w_xp    = (u16*)  take((size_t)(DTR + 2 * NST) * DINq * 2); //  0.4 MB
    u16*   w_dt    = (u16*)  take((size_t)DINq * DTR * 2);             //  0.25 MB
    u16*   w_out   = (u16*)  take((size_t)DMq * DINq * 2);             //  4 MB
    float* xz      = (float*)take((size_t)MROWS * 2 * DINq * 4);       // 32 MB
    float* xc      = (float*)take((size_t)MROWS * DINq * 4);           // 16 MB
    u16*   xc_bf   = (u16*)  take((size_t)MROWS * DINq * 2);           //  8 MB
    float* xdbl    = (float*)take((size_t)MROWS * (DTR + 2 * NST) * 4);//  0.8 MB
    u16*   dtA_bf  = (u16*)  take((size_t)MROWS * DTR * 2);            //  0.25 MB
    float* dtbuf   = (float*)take((size_t)MROWS * DINq * 4);           // 16 MB
    u16*   y_bf    = (u16*)  take((size_t)MROWS * DINq * 2);           //  8 MB

    // 1) weights -> bf16
    {
        int n;
        n = 2 * DINq * DMq;           k_f32_to_bf16<<<(n + 255) / 256, 256, 0, stream>>>(in_w,  w_in,  n);
        n = (DTR + 2 * NST) * DINq;   k_f32_to_bf16<<<(n + 255) / 256, 256, 0, stream>>>(xp_w,  w_xp,  n);
        n = DINq * DTR;               k_f32_to_bf16<<<(n + 255) / 256, 256, 0, stream>>>(dt_w,  w_dt,  n);
        n = DMq * DINq;               k_f32_to_bf16<<<(n + 255) / 256, 256, 0, stream>>>(out_w, w_out, n);
    }

    // 2) layernorm -> bf16 activations
    k_layernorm_bf16<<<MROWS, 256, 0, stream>>>(x, norm_w, norm_b, h_bf);

    // 3) in_proj: xz[2048,4096] = h[2048,1024] * in_w^T
    {
        dim3 grid((2 * DINq + 127) / 128, (MROWS + 63) / 64);
        k_gemm_bf16_wmma<<<grid, 256, 0, stream>>>(h_bf, w_in, xz, MROWS, 2 * DINq, DMq, nullptr);
    }

    // 4) depthwise conv + SiLU
    {
        int n = Bq * Lq * DINq;
        k_conv_silu<<<(n + 255) / 256, 256, 0, stream>>>(xz, conv_w, conv_b, xc, xc_bf);
    }

    // 5) x_proj: x_dbl[2048,96] = xc[2048,2048] * xp_w^T
    {
        dim3 grid((DTR + 2 * NST + 127) / 128, (MROWS + 63) / 64);
        k_gemm_bf16_wmma<<<grid, 256, 0, stream>>>(xc_bf, w_xp, xdbl, MROWS, DTR + 2 * NST, DINq, nullptr);
    }

    // 6) dt_proj: dt[2048,2048] = x_dbl[:, :64] * dt_w^T, then softplus(+bias)
    {
        int n = MROWS * DTR;
        k_extract_dt<<<(n + 255) / 256, 256, 0, stream>>>(xdbl, dtA_bf);
        dim3 grid((DINq + 127) / 128, (MROWS + 63) / 64);
        k_gemm_bf16_wmma<<<grid, 256, 0, stream>>>(dtA_bf, w_dt, dtbuf, MROWS, DINq, DTR, nullptr);
        n = MROWS * DINq;
        k_dt_softplus<<<(n + 255) / 256, 256, 0, stream>>>(dtbuf, dt_b);
    }

    // 7) selective scan + gating -> bf16
    {
        dim3 grid(DINq / 256, Bq);
        k_selective_scan<<<grid, 256, 0, stream>>>(xdbl, dtbuf, xc, xz, A_log, D_skip, y_bf);
    }

    // 8) out_proj with fused residual: out = x + y * out_w^T
    {
        dim3 grid((DMq + 127) / 128, (MROWS + 63) / 64);
        k_gemm_bf16_wmma<<<grid, 256, 0, stream>>>(y_bf, w_out, out, MROWS, DMq, DINq, x);
    }
}
